// RDQuantLinear_24180665877009
// MI455X (gfx1250) — matrix operand order repaired and checked by reading, not compiled, not run on hardware
//
#include <hip/hip_runtime.h>

// ---------------------------------------------------------------------------
// RDQuantLinear for MI455X (gfx1250, wave32, WMMA + async global->LDS)
//   out[b,s,o] = sum_k x[b,s,k] * (q_weight[o,k]-128)*scale[o] + bias[o]
// GEMM: M=8192, N=11008, K=4096. Block tile 128x128, K-step 64, 8 waves (2x4),
// each wave 64x32 via 4x2 v_wmma_f32_16x16x32_f16. Double-buffered LDS filled
// with GLOBAL_LOAD_ASYNC_TO_LDS_B128 (ASYNCcnt) -> no staging VGPRs, no spills.
// ---------------------------------------------------------------------------

typedef _Float16 v16h __attribute__((ext_vector_type(16)));
typedef _Float16 v8h  __attribute__((ext_vector_type(8)));
typedef float    v8f  __attribute__((ext_vector_type(8)));
typedef int      v4i  __attribute__((ext_vector_type(4)));

typedef __attribute__((address_space(1))) v4i gv4i;  // global int4
typedef __attribute__((address_space(3))) v4i lv4i;  // LDS int4

union Frag16 { v16h f; v8h h[2]; };

#define M_DIM 8192
#define N_DIM 11008
#define K_DIM 4096
#define BM 128
#define BN 128
#define BK 64
#define LDT 72            // padded LDS row stride (halves); 144B rows, conflict-free
#define TILE_H (BM * LDT) // one LDS buffer in halves

// ---- async 16B global->LDS copy (per-lane addresses), ASYNCcnt tracked ----
__device__ inline void async_copy_b128(const _Float16* g, _Float16* l) {
#if __has_builtin(__builtin_amdgcn_global_load_async_to_lds_b128)
  __builtin_amdgcn_global_load_async_to_lds_b128(
      (gv4i*)(void*)g, (lv4i*)(void*)l, 0, 0);
#else
  unsigned lofs = (unsigned)(__UINTPTR_TYPE__)(lv4i*)(void*)l;
  asm volatile("global_load_async_to_lds_b128 %0, %1, off"
               :: "v"(lofs), "v"(g) : "memory");
#endif
}

__device__ inline void wait_async0() {
#if __has_builtin(__builtin_amdgcn_s_wait_asynccnt)
  __builtin_amdgcn_s_wait_asynccnt(0);
#else
  asm volatile("s_wait_asynccnt 0" ::: "memory");
#endif
}

// ---------------- prep: fp32 -> fp16 ----------------
__global__ __launch_bounds__(256) void cvt_x_f32_to_f16(const float* __restrict__ x,
                                                        _Float16* __restrict__ xh) {
  size_t i = ((size_t)blockIdx.x * 256 + threadIdx.x) * 8;
  const float4* p = (const float4*)(x + i);
  float4 a = p[0], b = p[1];
  v8h o;
  o[0] = (_Float16)a.x; o[1] = (_Float16)a.y; o[2] = (_Float16)a.z; o[3] = (_Float16)a.w;
  o[4] = (_Float16)b.x; o[5] = (_Float16)b.y; o[6] = (_Float16)b.z; o[7] = (_Float16)b.w;
  *(v8h*)(xh + i) = o;
}

// ---------------- prep: int32 (0..255) -> fp16 (q-128), exact ----------------
__global__ __launch_bounds__(256) void cvt_w_i32_to_f16(const int* __restrict__ q,
                                                        _Float16* __restrict__ wh) {
  size_t i = ((size_t)blockIdx.x * 256 + threadIdx.x) * 8;
  const int4* p = (const int4*)(q + i);
  int4 a = p[0], b = p[1];
  v8h o;
  o[0] = (_Float16)(a.x - 128); o[1] = (_Float16)(a.y - 128);
  o[2] = (_Float16)(a.z - 128); o[3] = (_Float16)(a.w - 128);
  o[4] = (_Float16)(b.x - 128); o[5] = (_Float16)(b.y - 128);
  o[6] = (_Float16)(b.z - 128); o[7] = (_Float16)(b.w - 128);
  *(v8h*)(wh + i) = o;
}

// ---------------- main GEMM ----------------
__global__ __launch_bounds__(256) void wmma_gemm_dq(const _Float16* __restrict__ Xh,
                                                    const _Float16* __restrict__ Wh,
                                                    const float* __restrict__ scale,
                                                    const float* __restrict__ bias,
                                                    float* __restrict__ out) {
  __shared__ _Float16 lA[2 * TILE_H];
  __shared__ _Float16 lB[2 * TILE_H];

  const int tid   = threadIdx.x;
  const int lane  = tid & 31;
  const int wave  = tid >> 5;      // 0..7
  const int waveM = wave & 1;      // 2 waves along M
  const int waveN = wave >> 1;     // 4 waves along N
  const int ln15  = lane & 15;
  const int hi    = lane >> 4;     // 0: lanes 0-15, 1: lanes 16-31

  const int blockN0 = blockIdx.x * BN;
  const int blockM0 = blockIdx.y * BM;

  const _Float16* gA = Xh + (size_t)blockM0 * K_DIM;
  const _Float16* gB = Wh + (size_t)blockN0 * K_DIM;

  // this thread's 4 chunks: 1024 chunks of 8 halves = 128 rows x 64 halves
  const int row0[4] = { (0 * 256 + tid) >> 3, (1 * 256 + tid) >> 3,
                        (2 * 256 + tid) >> 3, (3 * 256 + tid) >> 3 };
  const int col0[4] = { ((0 * 256 + tid) & 7) * 8, ((1 * 256 + tid) & 7) * 8,
                        ((2 * 256 + tid) & 7) * 8, ((3 * 256 + tid) & 7) * 8 };

  v8f acc[4][2] = {};

  // prologue: stage 0 -> buffer 0
#pragma unroll
  for (int i = 0; i < 4; ++i) {
    async_copy_b128(gA + (size_t)row0[i] * K_DIM + col0[i], &lA[row0[i] * LDT + col0[i]]);
    async_copy_b128(gB + (size_t)row0[i] * K_DIM + col0[i], &lB[row0[i] * LDT + col0[i]]);
  }
  wait_async0();
  __syncthreads();

  const int KSTEPS = K_DIM / BK;   // 64
  for (int ks = 0; ks < KSTEPS; ++ks) {
    const int curBuf = (ks & 1) * TILE_H;

    // issue async loads for next stage into the other buffer
    if (ks + 1 < KSTEPS) {
      const int nxtBuf = ((ks + 1) & 1) * TILE_H;
      const int k0 = (ks + 1) * BK;
#pragma unroll
      for (int i = 0; i < 4; ++i) {
        async_copy_b128(gA + (size_t)row0[i] * K_DIM + k0 + col0[i],
                        &lA[nxtBuf + row0[i] * LDT + col0[i]]);
        async_copy_b128(gB + (size_t)row0[i] * K_DIM + k0 + col0[i],
                        &lB[nxtBuf + row0[i] * LDT + col0[i]]);
      }
    }

    // compute: two 16x16x32 K-chunks from current buffer
#pragma unroll
    for (int kc = 0; kc < 2; ++kc) {
      const int kb = kc * 32;

      // A frags: lanes 0-15 hold K0-7 / K16-23, lanes 16-31 hold K8-15 / K24-31
      Frag16 afrag[4];
#pragma unroll
      for (int mt = 0; mt < 4; ++mt) {
        const int row = waveM * 64 + mt * 16 + ln15;
        const _Float16* p = &lA[curBuf + row * LDT + kb + hi * 8];
        afrag[mt].h[0] = *(const v8h*)p;
        afrag[mt].h[1] = *(const v8h*)(p + 16);
      }

      // B frags: lanes 0-15 hold K0-7 / K16-23, lanes 16-31 hold K8-15 / K24-31
      Frag16 bfrag[2];
#pragma unroll
      for (int nt = 0; nt < 2; ++nt) {
        const int col = waveN * 32 + nt * 16 + ln15;
        const _Float16* p = &lB[curBuf + col * LDT + kb + hi * 8];
        bfrag[nt].h[0] = *(const v8h*)p;
        bfrag[nt].h[1] = *(const v8h*)(p + 16);
      }

#pragma unroll
      for (int mt = 0; mt < 4; ++mt)
#pragma unroll
        for (int nt = 0; nt < 2; ++nt)
          acc[mt][nt] = __builtin_amdgcn_wmma_f32_16x16x32_f16(
              false, afrag[mt].f, false, bfrag[nt].f,
              (short)0, acc[mt][nt], false, false);
    }

    wait_async0();     // own async loads for stage ks+1 landed
    __syncthreads();   // everyone done reading curBuf and done loading nxtBuf
  }

  // -------- epilogue: out = acc*scale[col] + bias[col] --------
  // C/D layout: lanes 0-15 -> N=lane, VGPR r -> M=r ; lanes 16-31 -> M=r+8
#pragma unroll
  for (int nt = 0; nt < 2; ++nt) {
    const int col = blockN0 + waveN * 32 + nt * 16 + ln15;
    const float s = scale[col];
    const float b = bias[col];
#pragma unroll
    for (int mt = 0; mt < 4; ++mt) {
      const int rbase = blockM0 + waveM * 64 + mt * 16 + hi * 8;
#pragma unroll
      for (int r = 0; r < 8; ++r) {
        out[(size_t)(rbase + r) * N_DIM + col] = acc[mt][nt][r] * s + b;
      }
    }
  }
}

// ---------------------------------------------------------------------------
extern "C" void kernel_launch(void* const* d_in, const int* in_sizes, int n_in,
                              void* d_out, int out_size, void* d_ws, size_t ws_size,
                              hipStream_t stream) {
  const float* x     = (const float*)d_in[0];  // [4,2048,4096]
  const int*   qw    = (const int*)d_in[1];    // [11008,4096]
  const float* scale = (const float*)d_in[2];  // [11008,1]
  const float* bias  = (const float*)d_in[3];  // [11008]
  float*       out   = (float*)d_out;          // [4,2048,11008]

  const size_t XH_BYTES = (size_t)M_DIM * K_DIM * sizeof(_Float16); // 64 MiB
  _Float16* Xh = (_Float16*)d_ws;
  _Float16* Wh = (_Float16*)((char*)d_ws + XH_BYTES);               // 86 MiB

  cvt_x_f32_to_f16<<<(M_DIM * (size_t)K_DIM) / 2048, 256, 0, stream>>>(x, Xh);
  cvt_w_i32_to_f16<<<(N_DIM * (size_t)K_DIM) / 2048, 256, 0, stream>>>(qw, Wh);

  dim3 grid(N_DIM / BN, M_DIM / BM);   // (86, 64)
  wmma_gemm_dq<<<grid, 256, 0, stream>>>(Xh, Wh, scale, bias, out);
}
